// RPN_56238301774304
// MI455X (gfx1250) — matrix operand order repaired and checked by reading, not compiled, not run on hardware
//
#include <hip/hip_runtime.h>
#include <stdint.h>

// ---------------- types ----------------
typedef __bf16 bf16_t;
typedef __attribute__((ext_vector_type(16))) __bf16 v16bf;
typedef __attribute__((ext_vector_type(8)))  __bf16 v8bf;
typedef __attribute__((ext_vector_type(8)))  float  v8f;
typedef __attribute__((ext_vector_type(4)))  unsigned int u32x4;
typedef __attribute__((ext_vector_type(8)))  int i32x8;
typedef __attribute__((ext_vector_type(4)))  int i32x4;

#if defined(__HIP_DEVICE_COMPILE__) && __has_builtin(__builtin_amdgcn_tensor_load_to_lds) && __has_builtin(__builtin_amdgcn_s_wait_tensorcnt)
#define USE_TDM 1
#else
#define USE_TDM 0
#endif

// ---------------- problem constants ----------------
#define NIMG 8
#define CIN  512
#define HH   64
#define WW   64
#define HP   66            // padded H/W
#define ANC  9
#define HWA  (HH*WW*ANC)   // 36864
#define PRE_K 2000
#define POST_K 300
#define NMS_TH 0.7f
#define NEGC  (-1e30f)
#define XCLIP 4.135166556742356f   // log(1000/16)

// ---------------- workspace layout (bytes, all 256-aligned) ----------------
#define WS_FEAT   ((size_t)0)                                      // bf16 [8][66][66][512]
#define WS_WCONV  (WS_FEAT  + (size_t)NIMG*HP*HP*CIN*2)            // bf16 [9][512][512]
#define WS_WHEAD  (WS_WCONV + (size_t)9*CIN*CIN*2)                 // bf16 [512][64]
#define WS_BHEAD  (WS_WHEAD + (size_t)CIN*64*2)                    // f32  [64]
#define WS_T      (WS_BHEAD + (size_t)256)                         // bf16 [32768][512]
#define WS_OBJ    (WS_T     + (size_t)NIMG*HH*WW*CIN*2)            // f32  [8][36864]
#define WS_DELTA  (WS_OBJ   + (size_t)NIMG*HWA*4)                  // f32  [8][36864*4]
#define WS_TIDX   (WS_DELTA + (size_t)NIMG*HWA*4*4)                // i32  [8][2048]
#define WS_TSC    (WS_TIDX  + (size_t)NIMG*2048*4)                 // f32  [8][2048]
#define WS_BOXES  (WS_TSC   + (size_t)NIMG*2048*4)                 // f32  [8][2000][4]

// base anchors = round(stack([-w,-h,w,h])/2) for sizes {128,256,512} x ratios {.5,1,2}
__constant__ float c_base[9][4] = {
  {-91.f,-45.f,91.f,45.f},   {-181.f,-91.f,181.f,91.f}, {-362.f,-181.f,362.f,181.f},
  {-64.f,-64.f,64.f,64.f},   {-128.f,-128.f,128.f,128.f},{-256.f,-256.f,256.f,256.f},
  {-45.f,-91.f,45.f,91.f},   {-91.f,-181.f,91.f,181.f}, {-181.f,-362.f,181.f,362.f}
};

// ---------------- TDM helper: 2D tile (bf16, data_size=2B) -> LDS ----------------
#if USE_TDM
__device__ __forceinline__ void tdm_load_2d(unsigned lds_byte_off, const void* gptr,
                                            unsigned tile0, unsigned tile1,
                                            unsigned tensor0, unsigned long long stride0,
                                            unsigned tensor1)
{
  unsigned long long ga = (unsigned long long)(uintptr_t)gptr;
  union { unsigned long long q[2]; u32x4 v; } g0;
  g0.q[0] = 1ull | ((unsigned long long)lds_byte_off << 32);        // count=1, lds_addr
  g0.q[1] = (ga & 0x01FFFFFFFFFFFFFFull) | (2ull << 62);            // global_addr[56:0], type=2
  union { unsigned long long q[4]; i32x8 v; } g1;
  g1.q[0] = (1ull << 16)                                            // data_size=1 (2 bytes)
          | ((unsigned long long)(tensor0 & 0xFFFFu) << 48);        // tensor_dim0[15:0]
  g1.q[1] = ((unsigned long long)(tensor0 >> 16))                   // tensor_dim0[31:16]
          | ((unsigned long long)tensor1 << 16)                     // tensor_dim1
          | ((unsigned long long)(tile0 & 0xFFFFu) << 48);          // tile_dim0
  g1.q[2] = ((unsigned long long)(tile1 & 0xFFFFu))                 // tile_dim1 (tile_dim2=0)
          | ((stride0 & 0xFFFFFFFFull) << 32);                      // tensor_dim0_stride[31:0]
  g1.q[3] = ((stride0 >> 32) & 0xFFFFull);                          // stride[47:32]; dim1 stride unused (2D)
  i32x4 z4 = {0,0,0,0};
#if __clang_major__ >= 23
  i32x8 z8 = {0,0,0,0,0,0,0,0};
  __builtin_amdgcn_tensor_load_to_lds(g0.v, g1.v, z4, z4, z8, 0);
#else
  __builtin_amdgcn_tensor_load_to_lds(g0.v, g1.v, z4, z4, 0);
#endif
}
#endif

// build v16bf fragment from two 16-byte LDS chunks
__device__ __forceinline__ v16bf ld_frag(const bf16_t* p0, const bf16_t* p1)
{
  v8bf lo = *(const v8bf*)p0;
  v8bf hi = *(const v8bf*)p1;
  v16bf r;
#pragma unroll
  for (int i = 0; i < 8; ++i) { r[i] = lo[i]; r[i+8] = hi[i]; }
  return r;
}

// ================= prep kernels (fp32 -> bf16, layout transforms) =================
__global__ void prep_pad_kernel(const float* __restrict__ feat, bf16_t* __restrict__ out)
{
  size_t g = (size_t)blockIdx.x*256 + threadIdx.x;          // over 8*66*66*512
  if (g >= (size_t)NIMG*HP*HP*CIN) return;
  int c  = (int)(g & (CIN-1));
  size_t r = g >> 9;
  int xp = (int)(r % HP); r /= HP;
  int yp = (int)(r % HP); int n = (int)(r / HP);
  float v = 0.f;
  int x = xp - 1, y = yp - 1;
  if (x >= 0 && x < WW && y >= 0 && y < HH)
    v = feat[(((size_t)n*CIN + c)*HH + y)*WW + x];
  out[g] = (bf16_t)v;
}

__global__ void prep_wconv_kernel(const float* __restrict__ w, bf16_t* __restrict__ wpk)
{
  size_t g = (size_t)blockIdx.x*256 + threadIdx.x;          // over 9*512*512 as [k][c][o]
  if (g >= (size_t)9*CIN*CIN) return;
  int o = (int)(g & (CIN-1));
  size_t r = g >> 9;
  int c = (int)(r % CIN);
  int k = (int)(r / CIN);
  wpk[g] = (bf16_t)w[((size_t)o*CIN + c)*9 + k];
}

__global__ void prep_whead_kernel(const float* __restrict__ cls_w, const float* __restrict__ bbox_w,
                                  const float* __restrict__ cls_b, const float* __restrict__ bbox_b,
                                  bf16_t* __restrict__ hw, float* __restrict__ hb)
{
  int g = blockIdx.x*256 + threadIdx.x;                     // over 512*64 as [c][j]
  if (g < CIN*64) {
    int j = g & 63, c = g >> 6;
    float v = 0.f;
    if (j < 9)       v = cls_w[(size_t)j*CIN + c];
    else if (j < 45) v = bbox_w[(size_t)(j-9)*CIN + c];
    hw[(size_t)c*64 + j] = (bf16_t)v;
  }
  if (g < 64) hb[g] = (g < 9) ? cls_b[g] : ((g < 45) ? bbox_b[g-9] : 0.f);
}

// ================= conv 3x3 + bias + relu via WMMA bf16 =================
// block = 256 thr (8 waves). Block tile M=64 pixels x N=256 out channels, K=9*512.
// Wave tile 32x64 = 2 M-tiles x 4 N-tiles (8 f32 accumulators).
#if USE_TDM
__device__ __forceinline__ void conv_issue(const bf16_t* feat, const bf16_t* wpk,
                                           int n, int y, int o0, int kk,
                                           unsigned ldsA, unsigned ldsB)
{
  const int k = kk >> 4, c0 = (kk & 15) << 5;
  const int ky = k/3, kx = k - 3*ky;
  const bf16_t* ga = feat + (((size_t)(n*HP + (y+ky))*HP + kx)*CIN + c0);
  tdm_load_2d(ldsA, ga, 32u, 64u,  (unsigned)CIN, (unsigned long long)CIN, 1u<<30);
  const bf16_t* gb = wpk + ((size_t)(k*CIN + c0)*CIN + o0);
  tdm_load_2d(ldsB, gb, 256u, 32u, (unsigned)CIN, (unsigned long long)CIN, 9u*CIN);
}
#endif

__global__ __launch_bounds__(256) void conv3x3_wmma_kernel(
    const bf16_t* __restrict__ feat,   // [8][66][66][512] padded NHWC
    const bf16_t* __restrict__ wpk,    // [9][512][512]  (tap, cin, cout)
    const float*  __restrict__ conv_b, // [512]
    bf16_t* __restrict__ tout)         // [8*4096][512]
{
  __shared__ __align__(16) bf16_t smA[2][64*32];    // 8 KB
  __shared__ __align__(16) bf16_t smB[2][32*256];   // 32 KB
  const int n   = blockIdx.x >> 6;
  const int y   = blockIdx.x & 63;
  const int o0  = blockIdx.y << 8;    // 0 / 256
  const int tid = threadIdx.x, lane = tid & 31, wid = tid >> 5;
  const int wm = (wid & 1) << 5;      // 0 / 32
  const int wn = (wid >> 1) << 6;     // 0,64,128,192
  v8f acc[2][4] = {};
  const int KSTEPS = 9*(CIN/32);      // 144
#if USE_TDM
  const unsigned ldsA[2] = {(unsigned)(uintptr_t)&smA[0][0], (unsigned)(uintptr_t)&smA[1][0]};
  const unsigned ldsB[2] = {(unsigned)(uintptr_t)&smB[0][0], (unsigned)(uintptr_t)&smB[1][0]};
  if (wid == 0) conv_issue(feat, wpk, n, y, o0, 0, ldsA[0], ldsB[0]);
#endif
  for (int kk = 0; kk < KSTEPS; ++kk) {
#if USE_TDM
    const int buf = kk & 1;
    if (wid == 0) {
      if (kk + 1 < KSTEPS) {
        conv_issue(feat, wpk, n, y, o0, kk+1, ldsA[buf^1], ldsB[buf^1]);
        __builtin_amdgcn_s_wait_tensorcnt(2);   // current buffer's 2 TDM ops done
      } else {
        __builtin_amdgcn_s_wait_tensorcnt(0);
      }
    }
    __syncthreads();
#else
    const int buf = 0;
    __syncthreads();
    {
      const int k = kk >> 4, c0 = (kk & 15) << 5;
      const int ky = k/3, kx = k - 3*ky;
      const bf16_t* ga = feat + (((size_t)(n*HP + (y+ky))*HP + kx)*CIN + c0);
      for (int i2 = tid; i2 < 64*4; i2 += 256) {
        int row = i2 >> 2, sub = i2 & 3;
        *(v8bf*)&smA[0][row*32 + sub*8] = *(const v8bf*)&ga[(size_t)row*CIN + sub*8];
      }
      const bf16_t* gb = wpk + ((size_t)(k*CIN + c0)*CIN + o0);
      for (int i2 = tid; i2 < 32*32; i2 += 256) {
        int row = i2 >> 5, sub = i2 & 31;
        *(v8bf*)&smB[0][row*256 + sub*8] = *(const v8bf*)&gb[(size_t)row*CIN + sub*8];
      }
    }
    __syncthreads();
#endif
    // fragments per ISA 7.12.2 bf16 layouts
    const int arow = lane & 15;
    const int koff = (lane >> 4) << 3;
    v16bf afrag[2], bfrag[4];
#pragma unroll
    for (int mt = 0; mt < 2; ++mt) {
      const bf16_t* p = &smA[buf][(wm + mt*16 + arow)*32 + koff];
      afrag[mt] = ld_frag(p, p + 16);
    }
#pragma unroll
    for (int nt = 0; nt < 4; ++nt) {
      const bf16_t* p = &smB[buf][wn + nt*16 + (lane & 15)];
#pragma unroll
      for (int s = 0; s < 16; ++s) bfrag[nt][s] = p[(koff + (s & 7) + ((s >> 3) << 4))*256];
    }
#pragma unroll
    for (int mt = 0; mt < 2; ++mt)
#pragma unroll
      for (int nt = 0; nt < 4; ++nt)
        acc[mt][nt] = __builtin_amdgcn_wmma_f32_16x16x32_bf16(
            false, afrag[mt], false, bfrag[nt], (short)0, acc[mt][nt], false, false);
    __syncthreads();
  }
  // epilogue: + bias, relu, store bf16 NHWC
  const int r8   = (lane >> 4) << 3;
  const int ocol = o0 + wn + (lane & 15);
  float bias[4];
#pragma unroll
  for (int nt = 0; nt < 4; ++nt) bias[nt] = conv_b[ocol + nt*16];
  const size_t prow = ((size_t)n*HH + y)*WW;
#pragma unroll
  for (int mt = 0; mt < 2; ++mt)
#pragma unroll
    for (int nt = 0; nt < 4; ++nt) {
      const int o = ocol + nt*16;
#pragma unroll
      for (int r = 0; r < 8; ++r) {
        int x = wm + mt*16 + r + r8;
        float v = acc[mt][nt][r] + bias[nt];
        v = v > 0.f ? v : 0.f;
        tout[(prow + x)*CIN + o] = (bf16_t)v;
      }
    }
}

// ================= fused 1x1 heads via WMMA: [32768 x 64] = t[32768x512] @ hw[512x64] =================
#if USE_TDM
__device__ __forceinline__ void head_issue(const bf16_t* t, const bf16_t* hw,
                                           size_t p0, int kk, unsigned ldsA, unsigned ldsB)
{
  const int c0 = kk << 5;
  tdm_load_2d(ldsA, t + p0*CIN + c0, 32u, 128u, (unsigned)CIN, (unsigned long long)CIN, 1u<<30);
  tdm_load_2d(ldsB, hw + (size_t)c0*64, 64u, 32u, 64u, 64ull, (unsigned)CIN);
}
#endif

__global__ __launch_bounds__(256) void head_wmma_kernel(
    const bf16_t* __restrict__ t,   // [32768][512]
    const bf16_t* __restrict__ hw,  // [512][64] (cols: 9 cls, 36 bbox, 19 pad)
    const float*  __restrict__ hb,  // [64]
    float* __restrict__ logits,     // [8][9][64][64]
    float* __restrict__ obj,        // [8][36864]
    float* __restrict__ delta)      // [8][36864*4]
{
  __shared__ __align__(16) bf16_t smA[2][128*32];
  __shared__ __align__(16) bf16_t smB[2][32*64];
  const int tid = threadIdx.x, lane = tid & 31, wid = tid >> 5;
  const size_t p0 = (size_t)blockIdx.x * 128;
  v8f acc[4] = {};
  const int KSTEPS = CIN/32; // 16
#if USE_TDM
  const unsigned ldsA[2] = {(unsigned)(uintptr_t)&smA[0][0], (unsigned)(uintptr_t)&smA[1][0]};
  const unsigned ldsB[2] = {(unsigned)(uintptr_t)&smB[0][0], (unsigned)(uintptr_t)&smB[1][0]};
  if (wid == 0) head_issue(t, hw, p0, 0, ldsA[0], ldsB[0]);
#endif
  for (int kk = 0; kk < KSTEPS; ++kk) {
#if USE_TDM
    const int buf = kk & 1;
    if (wid == 0) {
      if (kk + 1 < KSTEPS) {
        head_issue(t, hw, p0, kk+1, ldsA[buf^1], ldsB[buf^1]);
        __builtin_amdgcn_s_wait_tensorcnt(2);
      } else {
        __builtin_amdgcn_s_wait_tensorcnt(0);
      }
    }
    __syncthreads();
#else
    const int buf = 0;
    __syncthreads();
    {
      const int c0 = kk << 5;
      const bf16_t* ga = t + p0*CIN + c0;
      for (int i2 = tid; i2 < 128*4; i2 += 256) {
        int row = i2 >> 2, sub = i2 & 3;
        *(v8bf*)&smA[0][row*32 + sub*8] = *(const v8bf*)&ga[(size_t)row*CIN + sub*8];
      }
      const bf16_t* gb = hw + (size_t)c0*64;
      for (int i2 = tid; i2 < 32*8; i2 += 256) {
        int row = i2 >> 3, sub = i2 & 7;
        *(v8bf*)&smB[0][row*64 + sub*8] = *(const v8bf*)&gb[(size_t)row*64 + sub*8];
      }
    }
    __syncthreads();
#endif
    const int arow = wid*16 + (lane & 15);
    const int koff = (lane >> 4) << 3;
    const bf16_t* ap = &smA[buf][arow*32 + koff];
    v16bf afrag = ld_frag(ap, ap + 16);
#pragma unroll
    for (int nt = 0; nt < 4; ++nt) {
      const bf16_t* bp = &smB[buf][nt*16 + (lane & 15)];
      v16bf bfrag;
#pragma unroll
      for (int s = 0; s < 16; ++s) bfrag[s] = bp[(koff + (s & 7) + ((s >> 3) << 4))*64];
      acc[nt] = __builtin_amdgcn_wmma_f32_16x16x32_bf16(
          false, afrag, false, bfrag, (short)0, acc[nt], false, false);
    }
    __syncthreads();
  }
  // epilogue: bias, scatter to logits (NCHW), obj=sigmoid (NHWA), deltas (NHWA4)
  const int r8 = (lane >> 4) << 3;
  const int oc = lane & 15;
#pragma unroll
  for (int nt = 0; nt < 4; ++nt) {
    const int o = nt*16 + oc;
    const float bb = hb[o];
#pragma unroll
    for (int r = 0; r < 8; ++r) {
      int pl = wid*16 + r + r8;
      size_t pg = p0 + pl;
      int n   = (int)(pg >> 12);
      int pix = (int)(pg & 4095);
      float v = acc[nt][r] + bb;
      if (o < 9) {
        logits[(((size_t)n*ANC + o) << 12) + pix] = v;
        obj[(size_t)n*HWA + (size_t)pix*ANC + o] = 1.f/(1.f + expf(-v));
      } else if (o < 45) {
        delta[(size_t)n*HWA*4 + (size_t)pix*36 + (o-9)] = v;
      }
    }
  }
}

// ================= top-2000 per image (histogram radix select on float bits) =================
__global__ __launch_bounds__(1024) void topk_kernel(const float* __restrict__ obj,
                                                    int* __restrict__ tidx, float* __restrict__ tsc)
{
  __shared__ int hist[4096];
  __shared__ int s_T, s_prev, s_hi, s_eq;
  const int i = blockIdx.x;
  const float* sc = obj + (size_t)i*HWA;
  for (int b = threadIdx.x; b < 4096; b += 1024) hist[b] = 0;
  __syncthreads();
  for (int j = threadIdx.x; j < HWA; j += 1024)
    atomicAdd(&hist[(__float_as_uint(sc[j]) >> 20) & 0xFFF], 1);
  __syncthreads();
  if (threadIdx.x == 0) {
    int cum = 0, b = 4095;
    for (; b > 0; --b) { if (cum + hist[b] >= PRE_K) break; cum += hist[b]; }
    s_T = b; s_prev = cum; s_hi = 0; s_eq = 0;
  }
  __syncthreads();
  const int T = s_T, prev = s_prev;
  for (int j = threadIdx.x; j < HWA; j += 1024) {
    float v = sc[j];
    int key = (int)((__float_as_uint(v) >> 20) & 0xFFF);
    int slot = -1;
    if (key > T) slot = atomicAdd(&s_hi, 1);          // exactly 'prev' of these
    else if (key == T) {
      int p = atomicAdd(&s_eq, 1);
      if (prev + p < PRE_K) slot = prev + p;
    }
    if (slot >= 0 && slot < PRE_K) {
      tidx[(size_t)i*2048 + slot] = j;
      tsc [(size_t)i*2048 + slot] = v;
    }
  }
}

// ================= decode + clip =================
__global__ void decode_kernel(const int* __restrict__ tidx, const float* __restrict__ deltas,
                              const int* __restrict__ imgw, const int* __restrict__ imgh,
                              float* __restrict__ boxes)
{
  int g = blockIdx.x*blockDim.x + threadIdx.x;
  if (g >= NIMG*PRE_K) return;
  int i = g / PRE_K, j = g - i*PRE_K;
  int idx = tidx[(size_t)i*2048 + j];
  int a   = idx % ANC;
  int pix = idx / ANC;
  int x = pix & (WW-1), y = pix >> 6;
  float sx = (float)(x*16), sy = (float)(y*16);
  float ax1 = sx + c_base[a][0], ay1 = sy + c_base[a][1];
  float ax2 = sx + c_base[a][2], ay2 = sy + c_base[a][3];
  float w = ax2-ax1, h = ay2-ay1;
  float cx = ax1 + 0.5f*w, cy = ay1 + 0.5f*h;
  const float* d = deltas + (size_t)i*HWA*4 + (size_t)pix*36 + a*4;
  float dx=d[0], dy=d[1];
  float dw=fminf(d[2], XCLIP), dh=fminf(d[3], XCLIP);
  float pcx = dx*w + cx, pcy = dy*h + cy;
  float pw = expf(dw)*w, ph = expf(dh)*h;
  float fw = (float)imgw[0], fh = (float)imgh[0];
  float x1 = fminf(fmaxf(pcx - 0.5f*pw, 0.f), fw);
  float y1 = fminf(fmaxf(pcy - 0.5f*ph, 0.f), fh);
  float x2 = fminf(fmaxf(pcx + 0.5f*pw, 0.f), fw);
  float y2 = fminf(fmaxf(pcy + 0.5f*ph, 0.f), fh);
  float* b = boxes + (size_t)g*4;
  b[0]=x1; b[1]=y1; b[2]=x2; b[3]=y2;
}

// ================= NMS: 300 serial rounds, parallel argmax+suppress over 2000 =================
__global__ __launch_bounds__(512) void nms_kernel(const float* __restrict__ boxes,
                                                  const float* __restrict__ tsc,
                                                  float* __restrict__ out_boxes,
                                                  float* __restrict__ out_scores)
{
  const int i = blockIdx.x, t = threadIdx.x;
  __shared__ float smax[512];
  __shared__ int   simax[512];
  __shared__ float sb[5];
  __shared__ int   sbi;
  float bx1[4], by1[4], bx2[4], by2[4], sc[4];
  bool  val[4];
#pragma unroll
  for (int k = 0; k < 4; ++k) {
    int j = t + 512*k;
    if (j < PRE_K) {
      const float* b = boxes + ((size_t)i*PRE_K + j)*4;
      bx1[k]=b[0]; by1[k]=b[1]; bx2[k]=b[2]; by2[k]=b[3];
      sc[k] = tsc[(size_t)i*2048 + j];
      val[k] = true;                      // MIN_SIZE=0 and clip preserves x2>=x1, y2>=y1
    } else { bx1[k]=by1[k]=bx2[k]=by2[k]=0.f; sc[k]=NEGC; val[k]=false; }
  }
  for (int it = 0; it < POST_K; ++it) {
    float m = NEGC; int mi = -1;
#pragma unroll
    for (int k = 0; k < 4; ++k)
      if (val[k] && sc[k] > m) { m = sc[k]; mi = t + 512*k; }
    smax[t] = m; simax[t] = mi;
    __syncthreads();
    for (int s = 256; s > 0; s >>= 1) {
      if (t < s && smax[t+s] > smax[t]) { smax[t] = smax[t+s]; simax[t] = simax[t+s]; }
      __syncthreads();
    }
    if (t == 0) {
      float bs = smax[0]; int bi = simax[0];
      bool ok = (bs > NEGC) && (bi >= 0);
      float q0=0.f,q1=0.f,q2=0.f,q3=0.f;
      if (ok) {
        const float* b = boxes + ((size_t)i*PRE_K + bi)*4;
        q0=b[0]; q1=b[1]; q2=b[2]; q3=b[3];
      }
      sb[0]=q0; sb[1]=q1; sb[2]=q2; sb[3]=q3; sb[4]=bs; sbi = bi;
      float* ob = out_boxes + ((size_t)i*POST_K + it)*4;
      ob[0]=ok?q0:0.f; ob[1]=ok?q1:0.f; ob[2]=ok?q2:0.f; ob[3]=ok?q3:0.f;
      out_scores[(size_t)i*POST_K + it] = ok ? bs : NEGC;
    }
    __syncthreads();
    const int bi = sbi;
    if (sb[4] > NEGC && bi >= 0) {
      float q0=sb[0], q1=sb[1], q2=sb[2], q3=sb[3];
      float areaQ = (q2-q0)*(q3-q1);
#pragma unroll
      for (int k = 0; k < 4; ++k) {
        if (!val[k]) continue;
        float lx = fmaxf(q0, bx1[k]), ly = fmaxf(q1, by1[k]);
        float rx = fminf(q2, bx2[k]), ry = fminf(q3, by2[k]);
        float iw = fmaxf(rx-lx, 0.f), ih = fmaxf(ry-ly, 0.f);
        float inter = iw*ih;
        float areaK = (bx2[k]-bx1[k])*(by2[k]-by1[k]);
        if (inter / (areaQ + areaK - inter + 1e-9f) > NMS_TH) val[k] = false;
      }
    }
#pragma unroll
    for (int k = 0; k < 4; ++k)
      if (t + 512*k == bi) val[k] = false;
    __syncthreads();
  }
}

// ================= launch =================
extern "C" void kernel_launch(void* const* d_in, const int* in_sizes, int n_in,
                              void* d_out, int out_size, void* d_ws, size_t ws_size,
                              hipStream_t stream)
{
  (void)in_sizes; (void)n_in; (void)out_size; (void)ws_size;
  const float* features = (const float*)d_in[0];
  const float* conv_w   = (const float*)d_in[1];
  const float* conv_b   = (const float*)d_in[2];
  const float* cls_w    = (const float*)d_in[3];
  const float* cls_b    = (const float*)d_in[4];
  const float* bbox_w   = (const float*)d_in[5];
  const float* bbox_b   = (const float*)d_in[6];
  const int*   img_w    = (const int*)d_in[7];
  const int*   img_h    = (const int*)d_in[8];

  float* out = (float*)d_out;
  float* out_boxes  = out;                 // 8*300*4
  float* out_scores = out + NIMG*POST_K*4; // 8*300
  float* out_logits = out + NIMG*POST_K*5; // 8*9*64*64

  char* ws = (char*)d_ws;
  bf16_t* feat_pad = (bf16_t*)(ws + WS_FEAT);
  bf16_t* wpk      = (bf16_t*)(ws + WS_WCONV);
  bf16_t* head_w   = (bf16_t*)(ws + WS_WHEAD);
  float*  head_b   = (float*) (ws + WS_BHEAD);
  bf16_t* t_ws     = (bf16_t*)(ws + WS_T);
  float*  obj_ws   = (float*) (ws + WS_OBJ);
  float*  delta_ws = (float*) (ws + WS_DELTA);
  int*    tidx_ws  = (int*)   (ws + WS_TIDX);
  float*  tsc_ws   = (float*) (ws + WS_TSC);
  float*  boxes_ws = (float*) (ws + WS_BOXES);

  prep_pad_kernel  <<<69696, 256, 0, stream>>>(features, feat_pad);
  prep_wconv_kernel<<<9216,  256, 0, stream>>>(conv_w, wpk);
  prep_whead_kernel<<<128,   256, 0, stream>>>(cls_w, bbox_w, cls_b, bbox_b, head_w, head_b);

  conv3x3_wmma_kernel<<<dim3(NIMG*HH, CIN/256), 256, 0, stream>>>(feat_pad, wpk, conv_b, t_ws);
  head_wmma_kernel   <<<256, 256, 0, stream>>>(t_ws, head_w, head_b, out_logits, obj_ws, delta_ws);

  topk_kernel  <<<NIMG, 1024, 0, stream>>>(obj_ws, tidx_ws, tsc_ws);
  decode_kernel<<<(NIMG*PRE_K + 255)/256, 256, 0, stream>>>(tidx_ws, delta_ws, img_w, img_h, boxes_ws);
  nms_kernel   <<<NIMG, 512, 0, stream>>>(boxes_ws, tsc_ws, out_boxes, out_scores);
}
